// SlotAttention_48636209660113
// MI455X (gfx1250) — matrix operand order repaired and checked by reading, not compiled, not run on hardware
//
#include <hip/hip_runtime.h>
#include <hip/hip_bf16.h>
#include <stdint.h>

typedef __attribute__((ext_vector_type(16))) _Float16 v16h;
typedef __attribute__((ext_vector_type(8)))  _Float16 v8h;
typedef __attribute__((ext_vector_type(8)))  float    v8f;

#define WMMA_F16(a,b,c) __builtin_amdgcn_wmma_f32_16x16x32_f16(false,(a),false,(b),(short)0,(c),false,false)

#define LOG8F 2.07944154167983575f

// A-operand (16x32 f16): per lane, one contiguous 16-half run at K = 16*half (two 16B loads).
__device__ inline v16h load_a16(const _Float16* p) {
  v8h lo = *(const v8h*)(p);
  v8h hi = *(const v8h*)(p + 8);
  v16h r;
#pragma unroll
  for (int i = 0; i < 8; i++) { r[i] = lo[i]; r[i + 8] = hi[i]; }
  return r;
}

__device__ inline float wave_sum32(float x) {
#pragma unroll
  for (int st = 16; st >= 1; st >>= 1) x += __shfl_xor(x, st, 32);
  return x;
}

// ---------------------------------------------------------------------------
// K0: transpose-convert Wk, Wv (E x D f32) -> WkT, WvT (D x E f16), contiguous
// ---------------------------------------------------------------------------
__global__ __launch_bounds__(256) void kw_transpose(
    const float* __restrict__ Wk, const float* __restrict__ Wv,
    _Float16* __restrict__ WkT, _Float16* __restrict__ WvT)
{
  int idx = blockIdx.x * 256 + threadIdx.x;     // 0..131071
  int mat = idx >> 16;
  int r = idx & 65535;
  int e = r >> 8, d = r & 255;
  const float* W = mat ? Wv : Wk;
  _Float16* T = mat ? WvT : WkT;
  T[(size_t)d * 256 + e] = (_Float16)W[(size_t)e * 256 + d];
}

// ---------------------------------------------------------------------------
// K1: LayerNorm(inputs) -> x_f16 (LDS), then k = x@Wk, v = x@Wv via WMMA.
//     Both weight matrices staged in dynamic LDS (bank-padded stride 272),
//     so the tile loop reads B via ds_load instead of hammering L2.
//     Writes: kf16 [B*N,256] row-major, vT [B,256,4096] transposed (f16),
//     alogit [B*N] = x @ wi_W + wi_b.
//     64 rows per block; 8 waves: wave = (rowtile[0..3], mat[k|v]).
//     Dynamic LDS layout: [0)         Wk  256 rows * 272 halves
//                         [139264)    Wv  256 rows * 272 halves
//                         [278528)    xs   64 rows * 264 halves
//     total 312320 bytes (<= 320KB WGP LDS).
// ---------------------------------------------------------------------------
#define K1_LDS_BYTES 312320

__global__ __launch_bounds__(256) void k1_lnproj(
    const float* __restrict__ inputs, const float* __restrict__ lns,
    const float* __restrict__ lnb, const float* __restrict__ wiW,
    const float* __restrict__ wib,
    const _Float16* __restrict__ Wboth,   // WkT immediately followed by WvT
    _Float16* __restrict__ kf16, _Float16* __restrict__ vT,
    float* __restrict__ alogit)
{
  extern __shared__ char smem[];
  _Float16* wlds = (_Float16*)smem;                  // 512 rows * 272 halves
  _Float16* xsb  = (_Float16*)(smem + 278528);       // 64 rows * 264 halves
#define XS(r,c) xsb[(r)*264 + (c)]

  const int row0 = blockIdx.x * 64;
  const int wave = threadIdx.x >> 5, lane = threadIdx.x & 31;
  const int c0 = lane * 8;

  // ---- stage both weight matrices into LDS (row stride 256 -> 272) ----
  // 512 rows * 32 16B-units per row = 16384 units, 64 per thread.
  for (int u = threadIdx.x; u < 16384; u += 256) {
    int rowg = u >> 5;           // global row 0..511 (Wk then Wv)
    int off  = u & 31;           // 16B unit within row
    uint4 w = ((const uint4*)Wboth)[u];
    *(uint4*)(wlds + (size_t)rowg * 272 + off * 8) = w;
  }

  // ---- LayerNorm + wi logit, write x_f16 tile ----
  float lw[8], lbb[8], ww[8];
#pragma unroll
  for (int j = 0; j < 8; j++) { lw[j] = lns[c0 + j]; lbb[j] = lnb[c0 + j]; ww[j] = wiW[c0 + j]; }

  for (int rr = 0; rr < 8; rr++) {
    int r = wave * 8 + rr;
    const float* src = inputs + (size_t)(row0 + r) * 256 + c0;
    float x[8];
    float4 p0 = ((const float4*)src)[0];
    float4 p1 = ((const float4*)src)[1];
    x[0]=p0.x; x[1]=p0.y; x[2]=p0.z; x[3]=p0.w;
    x[4]=p1.x; x[5]=p1.y; x[6]=p1.z; x[7]=p1.w;
    float s = 0.f, s2 = 0.f;
#pragma unroll
    for (int j = 0; j < 8; j++) { s += x[j]; s2 += x[j]*x[j]; }
    s = wave_sum32(s); s2 = wave_sum32(s2);
    float mean = s * (1.f/256.f);
    float var  = s2 * (1.f/256.f) - mean*mean;
    float inv  = rsqrtf(var + 1e-5f);
    float wl = 0.f;
#pragma unroll
    for (int j = 0; j < 8; j++) {
      float xn = (x[j]-mean)*inv*lw[j] + lbb[j];
      XS(r, c0 + j) = (_Float16)xn;
      wl += xn * ww[j];
    }
    wl = wave_sum32(wl);
    if (lane == 0) alogit[row0 + r] = wl + wib[0];
  }
  __syncthreads();

  // ---- GEMM: 16 d-tiles per wave; A from xs LDS, B from weight LDS ----
  const int rt = wave & 3, isv = wave >> 4 ? 0 : (wave >> 2);  // isv = wave>>2 (0..1)
  const int isel = wave >> 2;
  const int col16 = lane & 15, half = lane >> 4;
  const _Float16* WT = wlds + (size_t)isel * (256 * 272);
  const _Float16* Ap = &XS(rt*16 + col16, half*16);
  const int growbase = row0 + rt*16;
  (void)isv;

  for (int j0 = 0; j0 < 16; j0++) {
    int d = j0*16 + col16;
    const _Float16* Bp = WT + (size_t)d * 272 + half*16;
    v16h bb[8];
#pragma unroll
    for (int kc = 0; kc < 8; kc++) bb[kc] = *(const v16h*)(Bp + kc*32);
    v8f acc = {};
#pragma unroll
    for (int kc = 0; kc < 8; kc++) {
      v16h a = load_a16(Ap + kc*32);
      acc = WMMA_F16(a, bb[kc], acc);
    }
    if (!isel) {
#pragma unroll
      for (int r = 0; r < 8; r++)
        kf16[(size_t)(growbase + r + 8*half) * 256 + d] = (_Float16)acc[r];
    } else {
      int bidx = growbase >> 12;                  // batch
      int nloc = (growbase & 4095) + 8*half;      // 8 consecutive n per lane
      v8h pk;
#pragma unroll
      for (int r = 0; r < 8; r++) pk[r] = (_Float16)acc[r];
      *(v8h*)(vT + ((size_t)bidx * 256 + d) * 4096 + nloc) = pk;
    }
  }
#undef XS
}

// ---------------------------------------------------------------------------
// K2: per-batch log-softmax of alogit over N -> log_a (+log NS)
// ---------------------------------------------------------------------------
__global__ __launch_bounds__(1024) void k2_loga(
    const float* __restrict__ alogit, float* __restrict__ loga)
{
  const int b = blockIdx.x, tid = threadIdx.x;
  const int lane = tid & 31, wave = tid >> 5;
  __shared__ float red[32][2];
  __shared__ float outv[2];
  float x[4];
  float m = -1e30f, s = 0.f;
#pragma unroll
  for (int i = 0; i < 4; i++) {
    x[i] = alogit[b*4096 + tid + i*1024];
    float t = x[i];
    if (t > m) { s = s*__expf(m - t) + 1.f; m = t; } else s += __expf(t - m);
  }
#pragma unroll
  for (int st = 16; st >= 1; st >>= 1) {
    float m2 = __shfl_xor(m, st, 32), s2 = __shfl_xor(s, st, 32);
    float nm = fmaxf(m, m2);
    s = s*__expf(m - nm) + s2*__expf(m2 - nm); m = nm;
  }
  if (lane == 0) { red[wave][0] = m; red[wave][1] = s; }
  __syncthreads();
  if (wave == 0) {
    float mm = red[lane][0], ss = red[lane][1];
#pragma unroll
    for (int st = 16; st >= 1; st >>= 1) {
      float m2 = __shfl_xor(mm, st, 32), s2 = __shfl_xor(ss, st, 32);
      float nm = fmaxf(mm, m2);
      ss = ss*__expf(mm - nm) + s2*__expf(m2 - nm); mm = nm;
    }
    if (lane == 0) { outv[0] = mm; outv[1] = ss; }
  }
  __syncthreads();
  float lse = outv[0] + __logf(outv[1]);
#pragma unroll
  for (int i = 0; i < 4; i++) loga[b*4096 + tid + i*1024] = x[i] - lse + LOG8F;
}

// ---------------------------------------------------------------------------
// K2b: k_sq per row from kf16
// ---------------------------------------------------------------------------
__global__ __launch_bounds__(256) void k2b_ksq(
    const _Float16* __restrict__ kf16, float* __restrict__ ksq)
{
  int row = blockIdx.x * 8 + (threadIdx.x >> 5);
  int lane = threadIdx.x & 31;
  v8h h = *(const v8h*)(kf16 + (size_t)row * 256 + lane * 8);
  float s = 0.f;
#pragma unroll
  for (int j = 0; j < 8; j++) { float f = (float)h[j]; s += f * f; }
  s = wave_sum32(s);
  if (lane == 0) ksq[row] = s;
}

// ---------------------------------------------------------------------------
// K3: slot-side: sn = LN(slots); log_b = log_softmax(sn@ws_W)+log8;
//     q = sn@Wq -> qpad f16 [B,16,256] (rows 8..15 zero), qsq [B,8]
// ---------------------------------------------------------------------------
__global__ __launch_bounds__(256) void k3_slots(
    const float* __restrict__ slots, const float* __restrict__ lns,
    const float* __restrict__ lnb, const float* __restrict__ wsW,
    const float* __restrict__ wsb, const float* __restrict__ Wq,
    float* __restrict__ logb, float* __restrict__ qsq,
    _Float16* __restrict__ qpad)
{
  const int b = blockIdx.x;
  const int wave = threadIdx.x >> 5, lane = threadIdx.x & 31;
  __shared__ float sn[8][256];
  __shared__ float wslog[8];
  __shared__ float redq[8][8];
  {
    const int s = wave, c0 = lane * 8;
    const float* src = slots + ((size_t)b*8 + s) * 256 + c0;
    float x[8];
    float4 p0 = ((const float4*)src)[0];
    float4 p1 = ((const float4*)src)[1];
    x[0]=p0.x; x[1]=p0.y; x[2]=p0.z; x[3]=p0.w;
    x[4]=p1.x; x[5]=p1.y; x[6]=p1.z; x[7]=p1.w;
    float su = 0.f, sq = 0.f;
#pragma unroll
    for (int j = 0; j < 8; j++) { su += x[j]; sq += x[j]*x[j]; }
    su = wave_sum32(su); sq = wave_sum32(sq);
    float mean = su * (1.f/256.f);
    float var  = sq * (1.f/256.f) - mean*mean;
    float inv  = rsqrtf(var + 1e-5f);
    float wl = 0.f;
#pragma unroll
    for (int j = 0; j < 8; j++) {
      float xn = (x[j]-mean)*inv*lns[c0+j] + lnb[c0+j];
      sn[s][c0 + j] = xn;
      wl += xn * wsW[c0 + j];
    }
    wl = wave_sum32(wl);
    if (lane == 0) wslog[s] = wl + wsb[0];
  }
  __syncthreads();
  if (threadIdx.x < 8) {
    float mx = -1e30f;
    for (int s2 = 0; s2 < 8; s2++) mx = fmaxf(mx, wslog[s2]);
    float sm = 0.f;
    for (int s2 = 0; s2 < 8; s2++) sm += __expf(wslog[s2] - mx);
    logb[b*8 + threadIdx.x] = wslog[threadIdx.x] - (mx + __logf(sm)) + LOG8F;
  }
  // q: thread tid computes q[s][tid] for s = 0..7
  const int d = threadIdx.x;
  float qv[8];
#pragma unroll
  for (int o = 0; o < 8; o++) {
    float acc = 0.f;
    for (int e = 0; e < 256; e++) acc += sn[o][e] * Wq[(size_t)e*256 + d];
    qpad[((size_t)b*16 + o) * 256 + d] = (_Float16)acc;
    qv[o] = acc * acc;
  }
#pragma unroll
  for (int s = 0; s < 8; s++) {
    float t = wave_sum32(qv[s]);
    if (lane == 0) redq[wave][s] = t;
  }
  for (int idx = threadIdx.x; idx < 8*256; idx += 256)
    qpad[((size_t)b*16 + 8) * 256 + idx] = (_Float16)0.f;
  __syncthreads();
  if (threadIdx.x < 8) {
    float a = 0.f;
    for (int w = 0; w < 8; w++) a += redq[w][threadIdx.x];
    qsq[b*8 + threadIdx.x] = a;
  }
}

// ---------------------------------------------------------------------------
// K4: cost C[b,n,s] = sqrt(max(ksq + qsq - 2*k.q, 1e-12)) via WMMA (s pad 16)
//     All 8 A/B chunks batch-loaded (single clause, one wait, 8 WMMAs).
// ---------------------------------------------------------------------------
__global__ __launch_bounds__(256) void k4_cost(
    const _Float16* __restrict__ kf16, const _Float16* __restrict__ qpad,
    const float* __restrict__ ksq, const float* __restrict__ qsq,
    float* __restrict__ Cout)
{
  const int wave = threadIdx.x >> 5, lane = threadIdx.x & 31;
  const int tile = blockIdx.x * 8 + wave;     // 0..8191
  const int b = tile >> 8, nt = tile & 255;
  const int n0 = nt * 16;
  const int col = lane & 15, half = lane >> 4;
  const _Float16* Abase = kf16 + ((size_t)b*4096 + n0 + col) * 256 + half*16;
  const _Float16* Bbase = qpad + ((size_t)b*16 + col) * 256 + half*16;
  v16h aa[8], bb[8];
#pragma unroll
  for (int kc = 0; kc < 8; kc++) {
    aa[kc] = load_a16(Abase + kc*32);
    bb[kc] = *(const v16h*)(Bbase + kc*32);
  }
  v8f acc = {};
#pragma unroll
  for (int kc = 0; kc < 8; kc++) acc = WMMA_F16(aa[kc], bb[kc], acc);
  if (col < 8) {
    float qs = qsq[b*8 + col];
#pragma unroll
    for (int r = 0; r < 8; r++) {
      int n = n0 + r + 8*half;
      float d2 = ksq[b*4096 + n] + qs - 2.0f * acc[r];
      Cout[((size_t)b*4096 + n) * 8 + col] = sqrtf(fmaxf(d2, 1e-12f));
    }
  }
}

// ---------------------------------------------------------------------------
// K5: per-batch MESH (4 entropy-gradient steps through 5-iter log-Sinkhorn,
//     analytic reverse-mode) + final 10-iter warm Sinkhorn -> attn.
//     C and dC register-resident (8 rows/thread, 512 threads).
// ---------------------------------------------------------------------------
#define RPT 8

__device__ inline void block_lse8(float m[8], float s[8],
    float (*red)[8][2], float (*outr)[2], int lane, int wave, float lse[8])
{
#pragma unroll
  for (int st = 16; st >= 1; st >>= 1) {
#pragma unroll
    for (int c = 0; c < 8; c++) {
      float m2 = __shfl_xor(m[c], st, 32);
      float s2 = __shfl_xor(s[c], st, 32);
      float nm = fmaxf(m[c], m2);
      s[c] = s[c]*__expf(m[c]-nm) + s2*__expf(m2-nm);
      m[c] = nm;
    }
  }
  __syncthreads();
  if (lane == 0) {
#pragma unroll
    for (int c = 0; c < 8; c++) { red[wave][c][0] = m[c]; red[wave][c][1] = s[c]; }
  }
  __syncthreads();
  if (wave == 0) {
    int c = lane & 7, p = lane >> 3;
    float mm = red[p][c][0], ss = red[p][c][1];
#pragma unroll
    for (int w = 1; w < 4; w++) {
      float m2 = red[p + 4*w][c][0], s2 = red[p + 4*w][c][1];
      float nm = fmaxf(mm, m2);
      ss = ss*__expf(mm-nm) + s2*__expf(m2-nm); mm = nm;
    }
#pragma unroll
    for (int st = 8; st <= 16; st <<= 1) {
      float m2 = __shfl_xor(mm, st, 32);
      float s2 = __shfl_xor(ss, st, 32);
      float nm = fmaxf(mm, m2);
      ss = ss*__expf(mm-nm) + s2*__expf(m2-nm); mm = nm;
    }
    if (p == 0) { outr[c][0] = mm; outr[c][1] = ss; }
  }
  __syncthreads();
#pragma unroll
  for (int c = 0; c < 8; c++) lse[c] = outr[c][0] + __logf(outr[c][1]);
}

__device__ inline void block_sum8(float s[8],
    float (*red)[8][2], float (*outr)[2], int lane, int wave, float out[8])
{
#pragma unroll
  for (int st = 16; st >= 1; st >>= 1) {
#pragma unroll
    for (int c = 0; c < 8; c++) s[c] += __shfl_xor(s[c], st, 32);
  }
  __syncthreads();
  if (lane == 0) {
#pragma unroll
    for (int c = 0; c < 8; c++) red[wave][c][0] = s[c];
  }
  __syncthreads();
  if (wave == 0) {
    int c = lane & 7, p = lane >> 3;
    float ss = red[p][c][0] + red[p+4][c][0] + red[p+8][c][0] + red[p+12][c][0];
#pragma unroll
    for (int st = 8; st <= 16; st <<= 1) ss += __shfl_xor(ss, st, 32);
    if (p == 0) outr[c][0] = ss;
  }
  __syncthreads();
#pragma unroll
  for (int c = 0; c < 8; c++) out[c] = outr[c][0];
}

__global__ __launch_bounds__(512) void k5_mesh(
    const float* __restrict__ Cin, const float* __restrict__ loga,
    const float* __restrict__ logb, _Float16* __restrict__ attnT,
    float* __restrict__ outAttnT, int writeOut)
{
  const int b = blockIdx.x, tid = threadIdx.x;
  const int lane = tid & 31, wave = tid >> 5;
  __shared__ float red[16][8][2];
  __shared__ float outr[8][2];
  __shared__ float vh_s[5][8];

  float C_[RPT][8], dC_[RPT][8], uh[5][RPT], la[RPT], lb8[8];
#pragma unroll
  for (int s = 0; s < 8; s++) lb8[s] = logb[b*8 + s];
  const float* Cb = Cin + (size_t)b * 4096 * 8;
#pragma unroll
  for (int i = 0; i < RPT; i++) {
    int n = tid + i*512;
    la[i] = loga[b*4096 + n];
    float4 c0 = ((const float4*)(Cb + (size_t)n*8))[0];
    float4 c1 = ((const float4*)(Cb + (size_t)n*8))[1];
    C_[i][0]=c0.x; C_[i][1]=c0.y; C_[i][2]=c0.z; C_[i][3]=c0.w;
    C_[i][4]=c1.x; C_[i][5]=c1.y; C_[i][6]=c1.z; C_[i][7]=c1.w;
  }
  float v[8], u[RPT];

  for (int ms = 0; ms < 4; ms++) {
#pragma unroll
    for (int s = 0; s < 8; s++) v[s] = 0.f;
    // ---- forward 5 Sinkhorn iterations, record potentials ----
#pragma unroll
    for (int t = 0; t < 5; t++) {
#pragma unroll
      for (int i = 0; i < RPT; i++) {
        float mx = -1e30f;
#pragma unroll
        for (int s = 0; s < 8; s++) mx = fmaxf(mx, v[s] - C_[i][s]);
        float sm = 0.f;
#pragma unroll
        for (int s = 0; s < 8; s++) sm += __expf(v[s] - C_[i][s] - mx);
        uh[t][i] = la[i] - (mx + __logf(sm));
      }
      float m8[8], s8[8];
#pragma unroll
      for (int s = 0; s < 8; s++) { m8[s] = -1e30f; s8[s] = 0.f; }
#pragma unroll
      for (int i = 0; i < RPT; i++) {
#pragma unroll
        for (int s = 0; s < 8; s++) {
          float t2 = uh[t][i] - C_[i][s];
          if (t2 > m8[s]) { s8[s] = s8[s]*__expf(m8[s]-t2) + 1.f; m8[s] = t2; }
          else s8[s] += __expf(t2 - m8[s]);
        }
      }
      float lse[8];
      block_lse8(m8, s8, red, outr, lane, wave, lse);
#pragma unroll
      for (int s = 0; s < 8; s++) v[s] = lb8[s] - lse[s];
      if (tid == 0) { for (int s = 0; s < 8; s++) vh_s[t][s] = v[s]; }
    }
    // ---- entropy adjoint seed: G = dent/d(logitP) ----
    float du[RPT], colacc[8];
#pragma unroll
    for (int s = 0; s < 8; s++) colacc[s] = 0.f;
#pragma unroll
    for (int i = 0; i < RPT; i++) {
      float acc = 0.f;
#pragma unroll
      for (int s = 0; s < 8; s++) {
        float lp = uh[4][i] + v[s] - C_[i][s];
        float P = __expf(lp);
        float G = -(__logf(P + 1e-8f) + P / (P + 1e-8f)) * P;
        dC_[i][s] = -G;
        acc += G;
        colacc[s] += G;
      }
      du[i] = acc;
    }
    float dv[8];
    block_sum8(colacc, red, outr, lane, wave, dv);
    // ---- reverse through the 5 iterations ----
#pragma unroll
    for (int t = 4; t >= 0; t--) {
      // v_t = log_b - LSE_n(-C + u_t):  S_t = exp(-C + u_t - log_b + v_t)
#pragma unroll
      for (int i = 0; i < RPT; i++) {
        float acc = 0.f;
#pragma unroll
        for (int s = 0; s < 8; s++) {
          float S = __expf(uh[t][i] + vh_s[t][s] - lb8[s] - C_[i][s]);
          float g = dv[s] * S;
          dC_[i][s] += g;
          acc += g;
        }
        du[i] -= acc;
      }
      // u_t = log_a - LSE_s(-C + v_{t-1}):  R_t = exp(-C + v_{t-1} - log_a + u_t)
#pragma unroll
      for (int s = 0; s < 8; s++) colacc[s] = 0.f;
#pragma unroll
      for (int i = 0; i < RPT; i++) {
#pragma unroll
        for (int s = 0; s < 8; s++) {
          float vp = (t > 0) ? vh_s[t-1][s] : 0.f;
          float R = __expf(vp + uh[t][i] - la[i] - C_[i][s]);
          float g = du[i] * R;
          dC_[i][s] += g;
          colacc[s] += g;
        }
      }
      if (t > 0) {
        float tmp[8];
        block_sum8(colacc, red, outr, lane, wave, tmp);
#pragma unroll
        for (int s = 0; s < 8; s++) dv[s] = -tmp[s];
      }
    }
    // ---- gradient step on C ----
#pragma unroll
    for (int i = 0; i < RPT; i++)
#pragma unroll
      for (int s = 0; s < 8; s++) C_[i][s] -= 5.0f * dC_[i][s];
  }

  // ---- final Sinkhorn: 5 fresh + 5 warm == 10 straight iterations ----
#pragma unroll
  for (int s = 0; s < 8; s++) v[s] = 0.f;
  for (int t = 0; t < 10; t++) {
#pragma unroll
    for (int i = 0; i < RPT; i++) {
      float mx = -1e30f;
#pragma unroll
      for (int s = 0; s < 8; s++) mx = fmaxf(mx, v[s] - C_[i][s]);
      float sm = 0.f;
#pragma unroll
      for (int s = 0; s < 8; s++) sm += __expf(v[s] - C_[i][s] - mx);
      u[i] = la[i] - (mx + __logf(sm));
    }
    float m8[8], s8[8];
#pragma unroll
    for (int s = 0; s < 8; s++) { m8[s] = -1e30f; s8[s] = 0.f; }
#pragma unroll
    for (int i = 0; i < RPT; i++) {
#pragma unroll
      for (int s = 0; s < 8; s++) {
        float t2 = u[i] - C_[i][s];
        if (t2 > m8[s]) { s8[s] = s8[s]*__expf(m8[s]-t2) + 1.f; m8[s] = t2; }
        else s8[s] += __expf(t2 - m8[s]);
      }
    }
    float lse[8];
    block_lse8(m8, s8, red, outr, lane, wave, lse);
#pragma unroll
    for (int s = 0; s < 8; s++) v[s] = lb8[s] - lse[s];
  }
#pragma unroll
  for (int i = 0; i < RPT; i++) {
    int n = tid + i*512;
#pragma unroll
    for (int s = 0; s < 8; s++) {
      float A = __expf(u[i] + v[s] - C_[i][s]);
      attnT[((size_t)b*16 + s) * 4096 + n] = (_Float16)A;
      if (writeOut) outAttnT[((size_t)b*8 + s) * 4096 + n] = A;
    }
  }
  for (int idx = tid; idx < 8*4096; idx += 512)
    attnT[(size_t)b*16*4096 + 8*4096 + idx] = (_Float16)0.f;
}

// ---------------------------------------------------------------------------
// K6: updates[b,s,d] = sum_n attn[b,n,s] * v[b,n,d] via WMMA, split-K over
//     8 slices of n (512 each); batch-loaded groups of 4 chunks;
//     fp32 partials -> K6r reduce (fixed order => deterministic).
// ---------------------------------------------------------------------------
__global__ __launch_bounds__(256) void k6_updates(
    const _Float16* __restrict__ attnT, const _Float16* __restrict__ vT,
    float* __restrict__ updPart)
{
  const int b = blockIdx.x >> 3, slice = blockIdx.x & 7;
  const int wave = threadIdx.x >> 5, lane = threadIdx.x & 31;
  const int col = lane & 15, half = lane >> 4;
  const int nbase = slice * 512;                  // 16 k-chunks of 32
  const _Float16* Abase = attnT + ((size_t)b*16 + col) * 4096 + nbase + half*16;
  const int d0a = wave * 16, d0b = (wave + 8) * 16;
  const _Float16* B0 = vT + ((size_t)b*256 + d0a + col) * 4096 + nbase + half*16;
  const _Float16* B1 = vT + ((size_t)b*256 + d0b + col) * 4096 + nbase + half*16;
  __builtin_prefetch(Abase + 512, 0, 1);
  v8f acc0 = {}, acc1 = {};
#pragma unroll
  for (int g = 0; g < 4; g++) {
    v16h a4[4], b04[4], b14[4];
#pragma unroll
    for (int k = 0; k < 4; k++) {
      int kc = g*4 + k;
      a4[k]  = load_a16(Abase + kc*32);
      b04[k] = *(const v16h*)(B0 + kc*32);
      b14[k] = *(const v16h*)(B1 + kc*32);
    }
#pragma unroll
    for (int k = 0; k < 4; k++) {
      acc0 = WMMA_F16(a4[k], b04[k], acc0);
      acc1 = WMMA_F16(a4[k], b14[k], acc1);
    }
  }
  if (half == 0) {
    float* base = updPart + (size_t)slice * 65536 + (size_t)b * 2048;
#pragma unroll
    for (int r = 0; r < 8; r++) {
      base[r * 256 + d0a + col] = acc0[r];
      base[r * 256 + d0b + col] = acc1[r];
    }
  }
}

__global__ __launch_bounds__(256) void k6_reduce(
    const float* __restrict__ updPart, float* __restrict__ upd)
{
  int idx = blockIdx.x * 256 + threadIdx.x;       // 0..65535
  float s = 0.f;
#pragma unroll
  for (int p = 0; p < 8; p++) s += updPart[(size_t)p * 65536 + idx];
  upd[idx] = s;
}

// ---------------------------------------------------------------------------
// K7: GRU cell + LN + MLP residual; one block per (b,s) row.
// ---------------------------------------------------------------------------
__global__ __launch_bounds__(256) void k7_gru(
    const float* __restrict__ upd, float* __restrict__ slots,
    const float* __restrict__ Wih, const float* __restrict__ Whh,
    const float* __restrict__ bih, const float* __restrict__ bhh,
    const float* __restrict__ lns, const float* __restrict__ lnb,
    const float* __restrict__ W1, const float* __restrict__ b1,
    const float* __restrict__ W2, const float* __restrict__ b2,
    float* __restrict__ out0, float* __restrict__ out2, int finalIt)
{
  const int row = blockIdx.x, j = threadIdx.x;
  const int lane = j & 31, wave = j >> 5;
  __shared__ float su[256], ss[256], sb[256];
  __shared__ float redw[8][2];
  su[j] = upd[(size_t)row*256 + j];
  ss[j] = slots[(size_t)row*256 + j];
  __syncthreads();
  float gi[3], gh[3];
#pragma unroll
  for (int g = 0; g < 3; g++) {
    const float* wi = Wih + (size_t)(g*256 + j) * 256;
    const float* wh = Whh + (size_t)(g*256 + j) * 256;
    float a = 0.f, c = 0.f;
    for (int e = 0; e < 256; e++) { a += su[e]*wi[e]; c += ss[e]*wh[e]; }
    gi[g] = a + bih[g*256 + j];
    gh[g] = c + bhh[g*256 + j];
  }
  float r = 1.f/(1.f + __expf(-(gi[0] + gh[0])));
  float z = 1.f/(1.f + __expf(-(gi[1] + gh[1])));
  float nn = tanhf(gi[2] + r*gh[2]);
  float h = (1.f - z)*nn + z*ss[j];
  float s1 = wave_sum32(h);
  float s2 = wave_sum32(h*h);
  if (lane == 0) { redw[wave][0] = s1; redw[wave][1] = s2; }
  __syncthreads();
  float ts = 0.f, tq = 0.f;
#pragma unroll
  for (int w = 0; w < 8; w++) { ts += redw[w][0]; tq += redw[w][1]; }
  float mean = ts * (1.f/256.f);
  float var  = tq * (1.f/256.f) - mean*mean;
  float inv  = rsqrtf(var + 1e-5f);
  float hn = (h - mean)*inv*lns[j] + lnb[j];
  sb[j] = hn;
  __syncthreads();
  float a1 = 0.f;
  for (int e = 0; e < 256; e++) a1 += sb[e] * W1[(size_t)e*256 + j];
  a1 = fmaxf(a1 + b1[j], 0.f);
  __syncthreads();
  su[j] = a1;
  __syncthreads();
  float o = 0.f;
  for (int e = 0; e < 256; e++) o += su[e] * W2[(size_t)e*256 + j];
  o += b2[j] + h;
  slots[(size_t)row*256 + j] = o;
  if (finalIt) { out0[(size_t)row*256 + j] = o; out2[(size_t)row*256 + j] = o; }
}

// ---------------------------------------------------------------------------
// Host-side orchestration
// ---------------------------------------------------------------------------
extern "C" void kernel_launch(void* const* d_in, const int* in_sizes, int n_in,
                              void* d_out, int out_size, void* d_ws, size_t ws_size,
                              hipStream_t stream)
{
  (void)in_sizes; (void)n_in; (void)out_size; (void)ws_size;
  const float* inputs    = (const float*)d_in[0];
  const float* slots_init= (const float*)d_in[1];
  const float* ln_in_s   = (const float*)d_in[2];
  const float* ln_in_b   = (const float*)d_in[3];
  const float* ln_sl_s   = (const float*)d_in[4];
  const float* ln_sl_b   = (const float*)d_in[5];
  const float* ln_mlp_s  = (const float*)d_in[6];
  const float* ln_mlp_b  = (const float*)d_in[7];
  const float* Wq        = (const float*)d_in[8];
  const float* Wk        = (const float*)d_in[9];
  const float* Wv        = (const float*)d_in[10];
  const float* wiW       = (const float*)d_in[11];
  const float* wib       = (const float*)d_in[12];
  const float* wsW       = (const float*)d_in[13];
  const float* wsb       = (const float*)d_in[14];
  const float* gWih      = (const float*)d_in[15];
  const float* gWhh      = (const float*)d_in[16];
  const float* gbih      = (const float*)d_in[17];
  const float* gbhh      = (const float*)d_in[18];
  const float* mW1       = (const float*)d_in[19];
  const float* mb1       = (const float*)d_in[20];
  const float* mW2       = (const float*)d_in[21];
  const float* mb2       = (const float*)d_in[22];

  char* ws = (char*)d_ws;
  _Float16* WkT   = (_Float16*)(ws + 0);        // contiguous with WvT below
  _Float16* WvT   = (_Float16*)(ws + 131072);
  _Float16* qpad  = (_Float16*)(ws + 262144);
  float*    ksq   = (float*)(ws + 524288);
  float*    loga  = (float*)(ws + 1048576);
  float*    alog  = (float*)(ws + 1572864);
  float*    logb  = (float*)(ws + 2097152);
  float*    qsq   = (float*)(ws + 2101248);
  float*    slots = (float*)(ws + 2105344);
  float*    upd   = (float*)(ws + 2367488);
  float*    Cbuf  = (float*)(ws + 2629632);
  _Float16* attnT = (_Float16*)(ws + 6823936);
  _Float16* kf16  = (_Float16*)(ws + 11018240);
  _Float16* vT    = (_Float16*)(ws + 78127104);
  float*    updP  = (float*)(ws + 145235968);
  // total workspace use: 147333120 bytes (~140.5 MiB)

  float* out        = (float*)d_out;
  float* out_slots  = out;
  float* out_attnT  = out + 65536;
  float* out_slots2 = out + 65536 + 1048576;

  kw_transpose<<<512, 256, 0, stream>>>(Wk, Wv, WkT, WvT);
  hipMemcpyAsync(slots, slots_init, (size_t)65536 * sizeof(float),
                 hipMemcpyDeviceToDevice, stream);
  k1_lnproj<<<2048, 256, K1_LDS_BYTES, stream>>>(inputs, ln_in_s, ln_in_b, wiW, wib,
                                                 WkT, kf16, vT, alog);
  k2_loga<<<32, 1024, 0, stream>>>(alog, loga);
  k2b_ksq<<<16384, 256, 0, stream>>>(kf16, ksq);

  for (int it = 0; it < 3; ++it) {
    int fin = (it == 2) ? 1 : 0;
    k3_slots<<<32, 256, 0, stream>>>(slots, ln_sl_s, ln_sl_b, wsW, wsb, Wq,
                                     logb, qsq, qpad);
    k4_cost<<<1024, 256, 0, stream>>>(kf16, qpad, ksq, qsq, Cbuf);
    k5_mesh<<<32, 512, 0, stream>>>(Cbuf, loga, logb, attnT, out_attnT, fin);
    k6_updates<<<256, 256, 0, stream>>>(attnT, vT, updP);
    k6_reduce<<<256, 256, 0, stream>>>(updP, upd);
    k7_gru<<<256, 256, 0, stream>>>(upd, slots, gWih, gWhh, gbih, gbhh,
                                    ln_mlp_s, ln_mlp_b, mW1, mb1, mW2, mb2,
                                    out_slots, out_slots2, fin);
  }
}
